// CharRNN_13752485281996
// MI455X (gfx1250) — matrix operand order repaired and checked by reading, not compiled, not run on hardware
//
#include <hip/hip_runtime.h>
#include <hip/hip_bf16.h>

// ---------------------------------------------------------------------------
// CharRNN on gfx1250: bf16 WMMA for all three matmul phases.
//   Phase 1: x_proj[S,H] = gather(emb)[S,E] @ Wax^T + bax     (WMMA GEMM)
//   Phase 2: sequential Elman recurrence, Waa resident in LDS (WMMA matvec)
//   Phase 3: ys[S,V]     = hs[S,H] @ Wya^T + bya              (WMMA GEMM)
// ---------------------------------------------------------------------------

#define SEQ  8192
#define EMB  256
#define HID  2048
#define NCH  256

typedef __attribute__((ext_vector_type(16))) __bf16        v16bf;
typedef __attribute__((ext_vector_type(8)))  float         v8f;
typedef __attribute__((ext_vector_type(4)))  unsigned int  u32x4;
typedef __attribute__((vector_size(16)))     int           vi4;   // builtin's pointee type

union ABu { v16bf v; u32x4 q[2]; };

// ---- CDNA5 async global->LDS availability probe ---------------------------
#if __has_builtin(__builtin_amdgcn_global_load_async_to_lds_b128) && \
    __has_builtin(__builtin_amdgcn_s_wait_asynccnt)
#define HAVE_ASYNC_LDS 1
#pragma message("CDNA5 probe: global_load_async_to_lds_b128 builtins AVAILABLE")
#else
#define HAVE_ASYNC_LDS 0
#pragma message("CDNA5 probe: global_load_async_to_lds_b128 builtins NOT available")
#endif

#if HAVE_ASYNC_LDS
// Builtin signature (from compiler diagnostic): param0 = vi4 __device__*
// (addrspace(1)), param1 = vi4 __shared__* (addrspace(3)), then offset, cpol.
__device__ __forceinline__ __attribute__((address_space(1))) vi4*
to_glb(const void* p) {
    // global generic pointers are bitwise identical to addrspace(1)
    return (__attribute__((address_space(1))) vi4*)(unsigned long long)p;
}
__device__ __forceinline__ __attribute__((address_space(3))) vi4*
to_lds(void* p) {
    // generic LDS pointer keeps the LDS byte offset in its low 32 bits
    return (__attribute__((address_space(3))) vi4*)(unsigned int)(unsigned long long)p;
}
#endif

__device__ __forceinline__ unsigned short f2bf(float x) {
    unsigned u = __builtin_bit_cast(unsigned, x);
    unsigned r = u + 0x7FFFu + ((u >> 16) & 1u);   // round-to-nearest-even
    return (unsigned short)(r >> 16);
}

__device__ __forceinline__ v8f vzero() {
    v8f z;
#pragma unroll
    for (int i = 0; i < 8; ++i) z[i] = 0.0f;
    return z;
}

// --------------------------- small utility kernels -------------------------

__global__ void k_cvt_bf16(const float* __restrict__ s, unsigned short* __restrict__ d, int n) {
    int i = blockIdx.x * blockDim.x + threadIdx.x;
    if (i < n) d[i] = f2bf(s[i]);
}

__global__ void k_embed(const int* __restrict__ seq, const float* __restrict__ tab,
                        unsigned short* __restrict__ d) {
    int i = blockIdx.x * blockDim.x + threadIdx.x;   // < SEQ*EMB
    int srow = i >> 8, e = i & 255;
    d[i] = f2bf(tab[seq[srow] * EMB + e]);
}

__global__ void k_init(unsigned* __restrict__ bar, unsigned short* __restrict__ h0,
                       unsigned short* __restrict__ h1) {
    if (threadIdx.x == 0) *bar = 0u;
    for (int j = threadIdx.x; j < HID; j += blockDim.x) { h0[j] = 0; h1[j] = 0; }
}

// ------------------------------ WMMA GEMM ----------------------------------
// C[M,N] = A[M,K](bf16,row-major) @ Bt[N,K](bf16,row-major == B^T) + bias[N]
// One wave computes a 16(M) x 64(N) tile; 8 waves / 256-thread block.
// A fragment (16-bit A 16x32): lane<16 -> row m, K = kb+{0..7,16..23};
// lane>=16 -> K = kb+{8..15,24..31}   => two contiguous 16B loads per lane.
// B fragment (16-bit B 32x16): lane<16 -> n=lane, K=kb+{0..7,16..23};
// lane>=16 -> n=lane-16, K=kb+{8..15,24..31}  => two contiguous 16B loads from Bt.
// Inner loop is software-pipelined (double-buffered fragments) so the 4
// WMMAs of step k overlap the global loads of step k+32.
__global__ void __launch_bounds__(256) k_gemm_bf16(
        const unsigned short* __restrict__ A,
        const unsigned short* __restrict__ Bt,
        const float* __restrict__ bias,
        float* __restrict__ C, int M, int N, int K) {
    int gw   = (blockIdx.x * 256 + (int)threadIdx.x) >> 5;
    int lane = threadIdx.x & 31;
    int tn   = N >> 6;                       // 64-wide wave tiles along N
    int mt   = gw / tn, nt = gw % tn;
    if (mt * 16 >= M) return;                // wave-uniform guard

    int half = lane >> 4;
    int m    = mt * 16 + (lane & 15);
    const unsigned short* Ar = A + (size_t)m * K;
    int akhl = half * 8;                     // A K-offset per lane half
    int bkhl = half * 8;                     // B K-offset per lane half

    const unsigned short* Brow[4];
#pragma unroll
    for (int s = 0; s < 4; ++s)
        Brow[s] = Bt + (size_t)(nt * 64 + s * 16 + (lane & 15)) * K + bkhl;

    v8f acc[4];
#pragma unroll
    for (int s = 0; s < 4; ++s) acc[s] = vzero();

    // prologue loads (kb = 0)
    ABu a0, b0[4];
    a0.q[0] = *(const u32x4*)(Ar + akhl);
    a0.q[1] = *(const u32x4*)(Ar + 16 + akhl);
#pragma unroll
    for (int s = 0; s < 4; ++s) {
        b0[s].q[0] = *(const u32x4*)(Brow[s]);
        b0[s].q[1] = *(const u32x4*)(Brow[s] + 16);
    }

#pragma unroll 2
    for (int kb = 0; kb < K; kb += 32) {
        int kn = (kb + 32 < K) ? (kb + 32) : kb;   // last iter: harmless reload
        ABu a1, b1[4];
        a1.q[0] = *(const u32x4*)(Ar + kn + akhl);
        a1.q[1] = *(const u32x4*)(Ar + kn + 16 + akhl);
#pragma unroll
        for (int s = 0; s < 4; ++s) {
            b1[s].q[0] = *(const u32x4*)(Brow[s] + kn);
            b1[s].q[1] = *(const u32x4*)(Brow[s] + kn + 16);
        }
#pragma unroll
        for (int s = 0; s < 4; ++s)
            acc[s] = __builtin_amdgcn_wmma_f32_16x16x32_bf16(
                         false, a0.v, false, b0[s].v, (short)0, acc[s], false, false);
        a0 = a1;
#pragma unroll
        for (int s = 0; s < 4; ++s) b0[s] = b1[s];
    }

    // D layout: VGPR r -> (M = mt*16 + half*8 + r, N = nt*64 + s*16 + lane%16)
    int rb = mt * 16 + half * 8;
#pragma unroll
    for (int s = 0; s < 4; ++s) {
        int col = nt * 64 + s * 16 + (lane & 15);
        float bv = bias[col];
#pragma unroll
        for (int r = 0; r < 8; ++r)
            C[(size_t)(rb + r) * N + col] = acc[s][r] + bv;
    }
}

// ----------------------------- recurrence ----------------------------------
// 32 persistent workgroups x 128 threads (4 waves). Each WG owns 64 rows of
// Waa, kept in LDS as bf16 (256KB). h ping-pongs between two global bf16
// buffers; a monotonic atomic counter is the per-step grid barrier.
// Matvec via WMMA: B tile = h replicated across all 16 N columns; column 0 of
// D (lanes 0 and 16) holds the true dot products for the wave's 16 rows.
// Global->LDS staging uses CDNA5 async-to-LDS (ASYNCcnt) when available.
__global__ void __launch_bounds__(128, 1) k_rnn(
        const float* __restrict__ xproj,             // [SEQ,HID] f32
        const unsigned short* __restrict__ WaaB,     // [HID,HID] bf16
        unsigned short* __restrict__ h0,             // [HID] bf16 ping
        unsigned short* __restrict__ h1,             // [HID] bf16 pong
        unsigned short* __restrict__ hs,             // [SEQ,HID] bf16
        float* __restrict__ hlast,                   // [HID] f32 (d_out tail)
        unsigned* __restrict__ bar) {
    extern __shared__ unsigned short lds[];
    unsigned short* w  = lds;                 // [64][HID] bf16
    unsigned short* hb = lds + 64 * HID;      // [HID]     bf16

    const int tid  = threadIdx.x;
    const int lane = tid & 31;
    const int wv   = tid >> 5;
    const int rowbase_wg = blockIdx.x * 64;
    const int nwg = gridDim.x;

    // one-time: stage this WG's 64-row Waa slice into LDS (bf16, 16B lanes)
    {
        const unsigned short* gsrc = WaaB + (size_t)rowbase_wg * HID;
#if HAVE_ASYNC_LDS
        for (int i = tid; i < 64 * HID / 8; i += 128)
            __builtin_amdgcn_global_load_async_to_lds_b128(
                to_glb(gsrc + (size_t)i * 8), to_lds(w + (size_t)i * 8), 0, 0);
        __builtin_amdgcn_s_wait_asynccnt(0);
#else
        for (int i = tid; i < 64 * HID / 8; i += 128)
            ((u32x4*)w)[i] = ((const u32x4*)gsrc)[i];
#endif
    }
    __syncthreads();

    const int rowbase = rowbase_wg + wv * 16;
    const int half  = lane >> 4;
    const int akhl  = half * 8;
    const int bkhl  = half * 8;
    const unsigned short* wrow = w + (size_t)(wv * 16 + (lane & 15)) * HID;

    for (int t = 0; t < SEQ; ++t) {
        const unsigned short* hin  = (t & 1) ? h1 : h0;
        unsigned short*       hout = (t & 1) ? h0 : h1;

        // stage h_prev into LDS (4KB): 256 x 16B
#if HAVE_ASYNC_LDS
        for (int i = tid; i < HID / 8; i += 128)
            __builtin_amdgcn_global_load_async_to_lds_b128(
                to_glb(hin + (size_t)i * 8), to_lds(hb + (size_t)i * 8), 0, 0);
        __builtin_amdgcn_s_wait_asynccnt(0);
#else
        for (int i = tid; i < HID / 8; i += 128)
            ((u32x4*)hb)[i] = ((const u32x4*)hin)[i];
#endif
        __syncthreads();

        // prefetch next step's x_proj slice while we do the matvec
        int rbpf = rowbase + (half ? 8 : 0);
        if ((lane == 0 || lane == 16) && t + 1 < SEQ)
            __builtin_prefetch(xproj + (size_t)(t + 1) * HID + rbpf, 0, 0);

        v8f acc = vzero();
        for (int kb = 0; kb < HID; kb += 32) {
            ABu a, b;
            a.q[0] = *(const u32x4*)(wrow + kb + akhl);
            a.q[1] = *(const u32x4*)(wrow + kb + 16 + akhl);
            b.q[0] = *(const u32x4*)(hb + kb + bkhl);       // h replicated
            b.q[1] = *(const u32x4*)(hb + kb + bkhl + 16);  // across N lanes
            acc = __builtin_amdgcn_wmma_f32_16x16x32_bf16(
                      false, a.v, false, b.v, (short)0, acc, false, false);
        }

        // column 0 of D lives in lane 0 (rows rb..rb+7) and lane 16 (rb+8..)
        if (lane == 0 || lane == 16) {
            int rb = rbpf;
            const float* xp = xproj + (size_t)t * HID + rb;
#pragma unroll
            for (int r = 0; r < 8; ++r) {
                float v = tanhf(acc[r] + xp[r]);
                unsigned short bv = f2bf(v);
                hout[rb + r] = bv;
                hs[(size_t)t * HID + rb + r] = bv;
                if (t == SEQ - 1) hlast[rb + r] = v;
            }
        }

        // grid-wide barrier: release, arrive, spin, acquire
        __threadfence();
        __syncthreads();
        if (tid == 0) {
            unsigned tgt = (unsigned)nwg * (unsigned)(t + 1);
            atomicAdd(bar, 1u);
            while (__hip_atomic_load(bar, __ATOMIC_ACQUIRE,
                                     __HIP_MEMORY_SCOPE_AGENT) < tgt)
                __builtin_amdgcn_s_sleep(2);
        }
        __syncthreads();
        __threadfence();
    }
}

// ------------------------------- launcher ----------------------------------

extern "C" void kernel_launch(void* const* d_in, const int* in_sizes, int n_in,
                              void* d_out, int out_size, void* d_ws, size_t ws_size,
                              hipStream_t stream) {
    const int*   seq  = (const int*)  d_in[0];
    const float* emb  = (const float*)d_in[1];
    const float* Wax  = (const float*)d_in[2];
    const float* bax  = (const float*)d_in[3];
    const float* Waa  = (const float*)d_in[4];
    const float* Wya  = (const float*)d_in[5];
    const float* bya  = (const float*)d_in[6];

    float* ys    = (float*)d_out;                       // [SEQ, NCH]
    float* hlast = (float*)d_out + (size_t)SEQ * NCH;   // [HID]

    // workspace layout (bytes)
    char* ws = (char*)d_ws;
    size_t off = 0;
    float*          xproj = (float*)(ws + off);          off += (size_t)SEQ * HID * 4;  // 64MB
    unsigned short* hsB   = (unsigned short*)(ws + off); off += (size_t)SEQ * HID * 2;  // 32MB
    unsigned short* embB  = (unsigned short*)(ws + off); off += (size_t)SEQ * EMB * 2;  //  4MB
    unsigned short* WaxB  = (unsigned short*)(ws + off); off += (size_t)HID * EMB * 2;  //  1MB
    unsigned short* WyaB  = (unsigned short*)(ws + off); off += (size_t)NCH * HID * 2;  //  1MB
    unsigned short* WaaB  = (unsigned short*)(ws + off); off += (size_t)HID * HID * 2;  //  8MB
    unsigned short* h0    = (unsigned short*)(ws + off); off += (size_t)HID * 2;
    unsigned short* h1    = (unsigned short*)(ws + off); off += (size_t)HID * 2;
    unsigned*       bar   = (unsigned*)(ws + off);       off += 256;

    const int RNN_WGS  = 32;
    const int RNN_THR  = 128;
    const int RNN_LDS  = (64 * HID + HID) * 2;   // 256KB + 4KB
    (void)hipFuncSetAttribute((const void*)k_rnn,
                              hipFuncAttributeMaxDynamicSharedMemorySize, RNN_LDS);

    // 0) reset barrier + h0/h1 (ws is poisoned; must re-init every call)
    k_init<<<1, 256, 0, stream>>>(bar, h0, h1);

    // 1) bf16 conversions of weights
    k_cvt_bf16<<<(HID * EMB + 255) / 256, 256, 0, stream>>>(Wax, WaxB, HID * EMB);
    k_cvt_bf16<<<(NCH * HID + 255) / 256, 256, 0, stream>>>(Wya, WyaB, NCH * HID);
    k_cvt_bf16<<<(HID * HID + 255) / 256, 256, 0, stream>>>(Waa, WaaB, HID * HID);

    // 2) embedding gather -> bf16 A matrix [SEQ, EMB]
    k_embed<<<(SEQ * EMB + 255) / 256, 256, 0, stream>>>(seq, emb, embB);

    // 3) x_proj = embB @ WaxB^T + bax   (Bt == Wax row-major, [HID,EMB])
    {
        int waves = (SEQ / 16) * (HID / 64);
        k_gemm_bf16<<<waves / 8, 256, 0, stream>>>(embB, WaxB, bax, xproj,
                                                   SEQ, HID, EMB);
    }

    // 4) sequential recurrence (persistent grid, LDS-resident Waa)
    k_rnn<<<RNN_WGS, RNN_THR, RNN_LDS, stream>>>(xproj, WaaB, h0, h1, hsB,
                                                 hlast, bar);

    // 5) ys = hs @ WyaB^T + bya         (Bt == Wya row-major, [NCH,HID])
    {
        int waves = (SEQ / 16) * (NCH / 64);
        k_gemm_bf16<<<waves / 8, 256, 0, stream>>>(hsB, WyaB, bya, ys,
                                                   SEQ, NCH, HID);
    }
}
